// MultiHeadSelfAttention_12146167513559
// MI455X (gfx1250) — matrix operand order repaired and checked by reading, not compiled, not run on hardware
//
#include <hip/hip_runtime.h>

// ---------------- problem constants ----------------
#define BB   8
#define NN   1024
#define EE   768
#define HH   12
#define DD   64
#define E3   2304          // 3*EE
#define MM   8192          // BB*NN
#define SLD  1032          // padded LDS row stride (floats) for S tile

typedef __attribute__((ext_vector_type(16))) __bf16 v16bf;
typedef __attribute__((ext_vector_type(8)))  float  v8f;

union FragBF {               // one WMMA bf16 operand (8 VGPRs)
    v16bf          v;
    uint4          q[2];
    unsigned short us[16];
};

__device__ __forceinline__ unsigned short f2bf(float f) {
    union { float f; unsigned int u; } c; c.f = f;
    unsigned int u = c.u;
    unsigned int r = u + 0x7FFFu + ((u >> 16) & 1u);   // round-to-nearest-even
    return (unsigned short)(r >> 16);
}

// A fragment (16x32 bf16): lane = row m0+(lane&15); two contiguous 8-elt K
// chunks at k0+8*hi and k0+16+8*hi  (ISA 7.12.2, 16-bit A 16x32 layout)
__device__ __forceinline__ void load_a_frag(FragBF& f, const unsigned short* __restrict__ base,
                                            int ld, int m0, int k0, int lane) {
    int row = m0 + (lane & 15);
    int hi  = lane >> 4;
    const unsigned short* p = base + row * ld + k0 + 8 * hi;
    f.q[0] = *reinterpret_cast<const uint4*>(p);
    f.q[1] = *reinterpret_cast<const uint4*>(p + 16);
}

// B fragment (32x16 bf16): lane = col n0+(lane&15); two contiguous 8-elt K
// chunks at k0+8*hi and k0+16+8*hi  (B stored [n][k] row-major, i.e. transposed weights)
__device__ __forceinline__ void load_b_frag(FragBF& f, const unsigned short* __restrict__ base,
                                            int ld, int n0, int k0, int lane) {
    int col = n0 + (lane & 15);
    int hi  = lane >> 4;
    const unsigned short* p = base + col * ld + k0 + 8 * hi;
    f.q[0] = *reinterpret_cast<const uint4*>(p);
    f.q[1] = *reinterpret_cast<const uint4*>(p + 16);
}

__device__ __forceinline__ v8f wmma_bf16(const FragBF& a, const FragBF& b, v8f c) {
    return __builtin_amdgcn_wmma_f32_16x16x32_bf16(false, a.v, false, b.v, (short)0, c, false, false);
}

// Shared 32(M) x 64(N) per-wave GEMM core, K-loop step 32.
__device__ __forceinline__ void gemm_core(const unsigned short* __restrict__ A, int lda,
                                          const unsigned short* __restrict__ Bm, int ldb,
                                          int m0, int n0, int K, int lane, v8f acc[2][4]) {
    for (int k0 = 0; k0 < K; k0 += 32) {
        FragBF a0, a1;
        load_a_frag(a0, A, lda, m0,      k0, lane);
        load_a_frag(a1, A, lda, m0 + 16, k0, lane);
        // pull next k-slab toward L2/L0 (global_prefetch_b8)
        if (k0 + 32 < K) {
            __builtin_prefetch(A  + (m0 + (lane & 15)) * lda + k0 + 32, 0, 1);
            __builtin_prefetch(Bm + (n0 + (lane & 15)) * ldb + k0 + 32, 0, 1);
        }
#pragma unroll
        for (int j = 0; j < 4; ++j) {
            FragBF b;
            load_b_frag(b, Bm, ldb, n0 + 16 * j, k0, lane);
            acc[0][j] = wmma_bf16(a0, b, acc[0][j]);
            acc[1][j] = wmma_bf16(a1, b, acc[1][j]);
        }
    }
}

// ---------------- conversion kernels ----------------
__global__ void k_cvt_x(const float* __restrict__ x, unsigned short* __restrict__ xb, int n4) {
    int i = blockIdx.x * blockDim.x + threadIdx.x;
    if (i >= n4) return;
    float4 v = reinterpret_cast<const float4*>(x)[i];
    uint2 o;
    o.x = (unsigned int)f2bf(v.x) | ((unsigned int)f2bf(v.y) << 16);
    o.y = (unsigned int)f2bf(v.z) | ((unsigned int)f2bf(v.w) << 16);
    reinterpret_cast<uint2*>(xb)[i] = o;
}

// Wt[n*K + k] = bf16(W[k*Ncols + n])   (coalesced writes)
__global__ void k_transpose_bf16(const float* __restrict__ W, unsigned short* __restrict__ Wt,
                                 int K, int Ncols) {
    int i = blockIdx.x * blockDim.x + threadIdx.x;
    int total = K * Ncols;
    if (i >= total) return;
    int kk = i % K;
    int nn = i / K;
    Wt[i] = f2bf(W[kk * Ncols + nn]);
}

// ---------------- GEMM1: qkv = x @ W_qkv + b, scattered to Q/K/Vt ----------------
__global__ __launch_bounds__(256) void k_gemm_qkv(const unsigned short* __restrict__ xb,
                                                  const unsigned short* __restrict__ wq,
                                                  const float* __restrict__ bias,
                                                  unsigned short* __restrict__ Qb,
                                                  unsigned short* __restrict__ Kb,
                                                  unsigned short* __restrict__ Vt) {
    int lane = threadIdx.x & 31;
    int w    = threadIdx.x >> 5;
    int wm   = w & 3, wn = w >> 2;
    int m0 = blockIdx.x * 128 + wm * 32;
    int n0 = blockIdx.y * 128 + wn * 64;
    v8f acc[2][4];
    const v8f z = {0.f,0.f,0.f,0.f,0.f,0.f,0.f,0.f};
#pragma unroll
    for (int i = 0; i < 2; ++i)
#pragma unroll
        for (int j = 0; j < 4; ++j) acc[i][j] = z;

    gemm_core(xb, EE, wq, EE, m0, n0, EE, lane, acc);

    int lo = lane & 15, hi = lane >> 4;
#pragma unroll
    for (int j = 0; j < 4; ++j) {
        int n   = n0 + 16 * j + lo;
        float bs = bias[n];
        int sel = n / EE;
        int rem = n - sel * EE;
        int h   = rem >> 6;        // head
        int d   = rem & 63;        // dim in head
#pragma unroll
        for (int i = 0; i < 2; ++i) {
#pragma unroll
            for (int r = 0; r < 8; ++r) {
                int m    = m0 + 16 * i + r + 8 * hi;
                int bidx = m >> 10;
                int nrow = m & 1023;
                unsigned short v = f2bf(acc[i][j][r] + bs);
                int bh = bidx * HH + h;
                if (sel == 0)      Qb[(bh * NN + nrow) * DD + d] = v;
                else if (sel == 1) Kb[(bh * NN + nrow) * DD + d] = v;
                else               Vt[(bh * DD + d) * NN + nrow] = v;   // V transposed
            }
        }
    }
}

// ---------------- attention: S = QK^T*scale, softmax, O = P V ----------------
__global__ __launch_bounds__(128) void k_attn(const unsigned short* __restrict__ Qb,
                                              const unsigned short* __restrict__ Kb,
                                              const unsigned short* __restrict__ Vt,
                                              unsigned short* __restrict__ Ob) {
    extern __shared__ float smem_f[];
    float* S    = smem_f;                 // 64 x SLD
    float* invs = smem_f + 64 * SLD;      // 64

    int rt = blockIdx.x, h = blockIdx.y, b = blockIdx.z;
    int lane = threadIdx.x & 31, w = threadIdx.x >> 5;
    int lo = lane & 15, hi = lane >> 4;
    int bh = b * HH + h;

    const unsigned short* Qbase = Qb + (bh * NN + rt * 64 + w * 16) * DD;
    const unsigned short* Kbase = Kb + bh * NN * DD;
    const unsigned short* Vbase = Vt + bh * DD * NN;

    FragBF aq[2];
#pragma unroll
    for (int kc = 0; kc < 2; ++kc) load_a_frag(aq[kc], Qbase, DD, 0, kc * 32, lane);

    const float scale = 0.03608439182435161f;   // 1/sqrt(768)
    const v8f z = {0.f,0.f,0.f,0.f,0.f,0.f,0.f,0.f};

    // ---- phase 1: score tile rows -> LDS ----
    for (int jt = 0; jt < 64; ++jt) {
        int j0 = jt * 16;
        v8f s = z;
#pragma unroll
        for (int kc = 0; kc < 2; ++kc) {
            FragBF bk;
            load_b_frag(bk, Kbase, DD, j0, kc * 32, lane);
            s = wmma_bf16(aq[kc], bk, s);
        }
        float* sp = S + (w * 16 + 8 * hi) * SLD + j0 + lo;
#pragma unroll
        for (int r = 0; r < 8; ++r) sp[r * SLD] = s[r] * scale;
    }
    __syncthreads();

    // ---- phase 2: row softmax (exp stored, 1/sum kept) ----
    if (threadIdx.x < 64) {
        float* sr = S + threadIdx.x * SLD;
        float mx = -3.0e38f;
        for (int j = 0; j < NN; j += 4) {
            float4 v = *reinterpret_cast<float4*>(sr + j);
            mx = fmaxf(mx, fmaxf(fmaxf(v.x, v.y), fmaxf(v.z, v.w)));
        }
        float sum = 0.f;
        for (int j = 0; j < NN; j += 4) {
            float4 v = *reinterpret_cast<float4*>(sr + j);
            v.x = __expf(v.x - mx); v.y = __expf(v.y - mx);
            v.z = __expf(v.z - mx); v.w = __expf(v.w - mx);
            sum += v.x + v.y + v.z + v.w;
            *reinterpret_cast<float4*>(sr + j) = v;
        }
        invs[threadIdx.x] = 1.0f / sum;
    }
    __syncthreads();

    // ---- phase 3: O = P @ V ----
    v8f acc[4];
#pragma unroll
    for (int t = 0; t < 4; ++t) acc[t] = z;

    const float* srow = S + (w * 16 + lo) * SLD;
    float iv = invs[w * 16 + lo];
    for (int jt = 0; jt < 32; ++jt) {
        int j0 = jt * 32;
        FragBF a;
        const float* p1 = srow + j0 + 8 * hi;
        const float* p2 = srow + j0 + 16 + 8 * hi;
#pragma unroll
        for (int e = 0; e < 8; ++e) a.us[e]     = f2bf(p1[e] * iv);
#pragma unroll
        for (int e = 0; e < 8; ++e) a.us[8 + e] = f2bf(p2[e] * iv);
#pragma unroll
        for (int t = 0; t < 4; ++t) {
            FragBF bv;
            load_b_frag(bv, Vbase, NN, t * 16, j0, lane);
            acc[t] = wmma_bf16(a, bv, acc[t]);
        }
    }

#pragma unroll
    for (int t = 0; t < 4; ++t) {
        int col = h * 64 + t * 16 + lo;
#pragma unroll
        for (int r = 0; r < 8; ++r) {
            int n = rt * 64 + w * 16 + r + 8 * hi;
            Ob[(b * NN + n) * EE + col] = f2bf(acc[t][r]);
        }
    }
}

// ---------------- GEMM2: out = O @ W_fc + b_fc (f32 out) ----------------
__global__ __launch_bounds__(256) void k_gemm_fc(const unsigned short* __restrict__ Ob,
                                                 const unsigned short* __restrict__ wf,
                                                 const float* __restrict__ bias,
                                                 float* __restrict__ out) {
    int lane = threadIdx.x & 31;
    int w    = threadIdx.x >> 5;
    int wm   = w & 3, wn = w >> 2;
    int m0 = blockIdx.x * 128 + wm * 32;
    int n0 = blockIdx.y * 128 + wn * 64;
    v8f acc[2][4];
    const v8f z = {0.f,0.f,0.f,0.f,0.f,0.f,0.f,0.f};
#pragma unroll
    for (int i = 0; i < 2; ++i)
#pragma unroll
        for (int j = 0; j < 4; ++j) acc[i][j] = z;

    gemm_core(Ob, EE, wf, EE, m0, n0, EE, lane, acc);

    int lo = lane & 15, hi = lane >> 4;
#pragma unroll
    for (int j = 0; j < 4; ++j) {
        int col = n0 + 16 * j + lo;
        float bs = bias[col];
#pragma unroll
        for (int i = 0; i < 2; ++i) {
#pragma unroll
            for (int r = 0; r < 8; ++r) {
                int m = m0 + 16 * i + r + 8 * hi;
                out[m * EE + col] = acc[i][j][r] + bs;
            }
        }
    }
}

// ---------------- host launcher ----------------
extern "C" void kernel_launch(void* const* d_in, const int* in_sizes, int n_in,
                              void* d_out, int out_size, void* d_ws, size_t ws_size,
                              hipStream_t stream) {
    (void)in_sizes; (void)n_in; (void)out_size; (void)ws_size;
    const float* x     = (const float*)d_in[0];
    const float* W_qkv = (const float*)d_in[1];
    const float* b_qkv = (const float*)d_in[2];
    const float* W_fc  = (const float*)d_in[3];
    const float* b_fc  = (const float*)d_in[4];
    float* out = (float*)d_out;

    // workspace partition (all sizes multiple of 256 B)
    char* ws = (char*)d_ws;
    unsigned short* xb = (unsigned short*)ws;  ws += (size_t)MM * EE * 2;   // 12.6 MB
    unsigned short* wq = (unsigned short*)ws;  ws += (size_t)E3 * EE * 2;   //  3.5 MB
    unsigned short* wf = (unsigned short*)ws;  ws += (size_t)EE * EE * 2;   //  1.2 MB
    unsigned short* Qb = (unsigned short*)ws;  ws += (size_t)BB * HH * NN * DD * 2;
    unsigned short* Kb = (unsigned short*)ws;  ws += (size_t)BB * HH * NN * DD * 2;
    unsigned short* Vt = (unsigned short*)ws;  ws += (size_t)BB * HH * NN * DD * 2;
    unsigned short* Ob = (unsigned short*)ws;  ws += (size_t)MM * EE * 2;

    // 1) convert activations + transpose/convert weights to bf16
    {
        int n4 = MM * EE / 4;
        k_cvt_x<<<(n4 + 255) / 256, 256, 0, stream>>>(x, xb, n4);
        int t1 = EE * E3;
        k_transpose_bf16<<<(t1 + 255) / 256, 256, 0, stream>>>(W_qkv, wq, EE, E3);
        int t2 = EE * EE;
        k_transpose_bf16<<<(t2 + 255) / 256, 256, 0, stream>>>(W_fc, wf, EE, EE);
    }
    // 2) QKV projection
    k_gemm_qkv<<<dim3(MM / 128, E3 / 128), 256, 0, stream>>>(xb, wq, b_qkv, Qb, Kb, Vt);
    // 3) attention (264448 B dynamic LDS; CDNA5 WGP has 320 KB)
    {
        size_t smem = (size_t)64 * SLD * sizeof(float) + 64 * sizeof(float);
        k_attn<<<dim3(NN / 64, HH, BB), 128, smem, stream>>>(Qb, Kb, Vt, Ob);
    }
    // 4) output projection
    k_gemm_fc<<<dim3(MM / 128, EE / 128), 256, 0, stream>>>(Ob, wf, b_fc, out);
}
